// MultiHeadLatentAttention_44186623542098
// MI455X (gfx1250) — matrix operand order repaired and checked by reading, not compiled, not run on hardware
//
#include <hip/hip_runtime.h>

#define BB 2
#define TT 2048
#define DDIM 2048
#define HH 16
#define HDIM 128
#define KVHH 4
#define QLAT 1536
#define KVLAT 512

typedef unsigned short u16;
typedef unsigned int   u32;
typedef __attribute__((ext_vector_type(16))) __bf16       v16bf;
typedef __attribute__((ext_vector_type(8)))  float        v8f;
typedef __attribute__((ext_vector_type(8)))  unsigned int u32x8;
typedef int v4i_vs __attribute__((vector_size(16)));   // matches builtin param type

#define AS1 __attribute__((address_space(1)))
#define AS3 __attribute__((address_space(3)))

// ---------- async global->LDS copy (16 bytes), CDNA5 path ----------

__device__ inline void async_copy_b128(const u16* g, u16* l) {
#if __has_builtin(__builtin_amdgcn_global_load_async_to_lds_b128)
  __builtin_amdgcn_global_load_async_to_lds_b128(
      (AS1 v4i_vs*)(u16*)g, (AS3 v4i_vs*)l, 0, 0);
#elif defined(__gfx1250__)
  u32 lofs = (u32)(unsigned long long)(AS3 void*)l;   // LDS byte offset
  asm volatile("global_load_async_to_lds_b128 %0, %1, off"
               :: "v"(lofs), "v"(g) : "memory");
#else
  *(uint4*)l = *(const uint4*)g;
#endif
}

__device__ inline void async_wait_all() {
#if __has_builtin(__builtin_amdgcn_s_wait_asynccnt)
  __builtin_amdgcn_s_wait_asynccnt(0);
#elif defined(__gfx1250__)
  asm volatile("s_wait_asynccnt 0" ::: "memory");
#endif
}

// ---------- helpers ----------

__device__ inline u16 f2bf(float f) {
  union { float f; u32 u; } cv; cv.f = f;
  u32 u = cv.u;
  u32 r = u + 0x7FFFu + ((u >> 16) & 1u);   // round-to-nearest-even
  return (u16)(r >> 16);
}

// A-fragment (16x32 bf16, MxK): lane row M=l%16, K = 16*(l/16)+{0..15}.
__device__ inline v16bf load_a_frag(const u16* p, int rowStride, int lane) {
  const int l16 = lane & 15, lh = lane >> 4;
  const u32* q = (const u32*)(p + (size_t)l16 * rowStride + lh * 16);
  u32x8 r;
  r[0] = q[0];  r[1] = q[1];  r[2] = q[2];  r[3] = q[3];
  r[4] = q[4];  r[5] = q[5];  r[6] = q[6];  r[7] = q[7];
  return __builtin_bit_cast(v16bf, r);
}

// B-fragment (32x16 bf16, KxN) from N-major W[N][K]: lane n=l16, K = 16*(l/16)+{0..15}.
__device__ inline v16bf load_b_frag(const u16* p, int rowStride, int lane) {
  const int l16 = lane & 15, lh = lane >> 4;
  const u32* q = (const u32*)(p + (size_t)l16 * rowStride + lh * 16);
  u32x8 r;
#pragma unroll
  for (int i = 0; i < 8; ++i) r[i] = q[i];
  return __builtin_bit_cast(v16bf, r);
}

__device__ inline v8f wmma_bf16(v16bf a, v16bf b, v8f c) {
  return __builtin_amdgcn_wmma_f32_16x16x32_bf16(
      false, a, false, b, (short)0, c, false, false);
}

// ---------- elementwise kernels ----------

__global__ void cvt_f32_bf16(const float* __restrict__ in, u16* __restrict__ out, size_t n) {
  size_t i = (size_t)blockIdx.x * blockDim.x + threadIdx.x;
  if (i < n) out[i] = f2bf(in[i]);
}

// in: [R][C] -> out: [C][R]  (used for Wuk [128][512] -> [512][128])
__global__ void cvt_T_bf16(const float* __restrict__ in, u16* __restrict__ out, int R, int C) {
  size_t i = (size_t)blockIdx.x * blockDim.x + threadIdx.x;
  if (i >= (size_t)R * C) return;
  int c = (int)(i % C);
  int r = (int)(i / C);
  out[(size_t)c * R + r] = f2bf(in[i]);
}

__global__ __launch_bounds__(256) void rms_norm_bf16(
    const float* __restrict__ in, const float* __restrict__ w,
    u16* __restrict__ out, int C) {
  __shared__ float red[8];
  int row = blockIdx.x;
  const float* r = in + (size_t)row * C;
  float ss = 0.f;
  for (int i = threadIdx.x; i < C; i += 256) { float v = r[i]; ss += v * v; }
#pragma unroll
  for (int off = 16; off >= 1; off >>= 1) ss += __shfl_xor(ss, off, 32);
  int lane = threadIdx.x & 31, wv = threadIdx.x >> 5;
  if (lane == 0) red[wv] = ss;
  __syncthreads();
  if (wv == 0) {
    float t = (lane < 8) ? red[lane] : 0.f;
#pragma unroll
    for (int off = 4; off >= 1; off >>= 1) t += __shfl_xor(t, off, 32);
    if (lane == 0) red[0] = t;
  }
  __syncthreads();
  float inv = rsqrtf(red[0] / (float)C + 1e-6f);
  for (int i = threadIdx.x; i < C; i += 256)
    out[(size_t)row * C + i] = f2bf(r[i] * inv * w[i]);
}

// Qr rope: in/out [B,T,H,HD], angle indexed by t
__global__ void rope_t_bf16(const float* __restrict__ in, u16* __restrict__ out) {
  size_t idx = (size_t)blockIdx.x * blockDim.x + threadIdx.x;
  size_t total = (size_t)BB * TT * HH * (HDIM / 2);
  if (idx >= total) return;
  int i = (int)(idx % (HDIM / 2));
  size_t r = idx / (HDIM / 2);
  int t = (int)((r / HH) % TT);
  float inv = __powf(10000.f, -(2.f * i) / (float)HDIM);
  float ang = (float)t * inv;
  float cc = __cosf(ang), sn = __sinf(ang);
  size_t base = idx * 2;
  float x1 = in[base], x2 = in[base + 1];
  out[base]     = f2bf(x1 * cc - x2 * sn);
  out[base + 1] = f2bf(x1 * sn + x2 * cc);
}

// K rope: in [B,T,KVH,HD] -> out [B,KVH,T,HD]; angle indexed by kv-head g (per reference)
__global__ void rope_h_bf16(const float* __restrict__ in, u16* __restrict__ out) {
  size_t idx = (size_t)blockIdx.x * blockDim.x + threadIdx.x;
  size_t total = (size_t)BB * TT * KVHH * (HDIM / 2);
  if (idx >= total) return;
  int i = (int)(idx % (HDIM / 2));
  size_t r = idx / (HDIM / 2);
  int g = (int)(r % KVHH); r /= KVHH;
  int t = (int)(r % TT);
  int b = (int)(r / TT);
  float inv = __powf(10000.f, -(2.f * i) / (float)HDIM);
  float ang = (float)g * inv;
  float cc = __cosf(ang), sn = __sinf(ang);
  size_t ibase = idx * 2;
  size_t obase = (((size_t)(b * KVHH + g) * TT) + t) * HDIM + 2 * i;
  float x1 = in[ibase], x2 = in[ibase + 1];
  out[obase]     = f2bf(x1 * cc - x2 * sn);
  out[obase + 1] = f2bf(x1 * sn + x2 * cc);
}

// V transpose: in [B,T,KVH*HD] f32 -> out [B,KVH,HD,T] bf16
__global__ void v_trans_bf16(const float* __restrict__ in, u16* __restrict__ out) {
  size_t idx = (size_t)blockIdx.x * blockDim.x + threadIdx.x;
  size_t total = (size_t)BB * TT * KVLAT;
  if (idx >= total) return;
  int d = (int)(idx % HDIM);
  size_t r = idx / HDIM;
  int g = (int)(r % KVHH); r /= KVHH;
  int t = (int)(r % TT);
  int b = (int)(r / TT);
  out[(((size_t)(b * KVHH + g) * HDIM) + d) * TT + t] = f2bf(in[idx]);
}

// ---------- tiled bf16 WMMA GEMM:  C[M,N] = A[M,K] * W[N,K]^T ----------

#define BM 128
#define BN 128
#define BK 32

template <bool OUT_BF16>
__global__ __launch_bounds__(256) void gemm_bf16_nt(
    const u16* __restrict__ A, const u16* __restrict__ Bw,
    void* __restrict__ Cout, int M, int N, int K) {
  __shared__ u16 As[BM * BK];
  __shared__ u16 Bs[BN * BK];
  int tid  = threadIdx.x;
  int lane = tid & 31;
  int wid  = tid >> 5;
  int wm   = wid & 3;   // 4 waves along M (32 rows each)
  int wn   = wid >> 2;  // 2 waves along N (64 cols each)
  int bm = blockIdx.x, bn = blockIdx.y;
  size_t aBase = (size_t)bm * BM * K;
  size_t bBase = (size_t)bn * BN * K;

  v8f acc[2][4];
#pragma unroll
  for (int i = 0; i < 2; ++i)
#pragma unroll
    for (int j = 0; j < 4; ++j)
#pragma unroll
      for (int v = 0; v < 8; ++v) acc[i][j][v] = 0.f;

  for (int k0 = 0; k0 < K; k0 += BK) {
    __syncthreads();   // previous tile's readers done
#pragma unroll
    for (int i = tid; i < 512; i += 256) {   // 512 x b128 per tile
      int row = i >> 2, kq = (i & 3) * 8;
      async_copy_b128(A  + aBase + (size_t)row * K + k0 + kq, As + row * BK + kq);
      async_copy_b128(Bw + bBase + (size_t)row * K + k0 + kq, Bs + row * BK + kq);
    }
    if (k0 + BK < K) {  // L2 prefetch of next tile (global_prefetch_b8)
      __builtin_prefetch(A  + aBase + (size_t)(tid >> 1) * K + k0 + BK + (tid & 1) * 16, 0, 1);
      __builtin_prefetch(Bw + bBase + (size_t)(tid >> 1) * K + k0 + BK + (tid & 1) * 16, 0, 1);
    }
    async_wait_all();
    __syncthreads();
    v16bf af[2], bfr[4];
#pragma unroll
    for (int i = 0; i < 2; ++i) af[i] = load_a_frag(As + (wm * 32 + i * 16) * BK, BK, lane);
#pragma unroll
    for (int j = 0; j < 4; ++j) bfr[j] = load_b_frag(Bs + (wn * 64 + j * 16) * BK, BK, lane);
#pragma unroll
    for (int i = 0; i < 2; ++i)
#pragma unroll
      for (int j = 0; j < 4; ++j)
        acc[i][j] = wmma_bf16(af[i], bfr[j], acc[i][j]);
  }

  int l16 = lane & 15, lh = lane >> 4;
#pragma unroll
  for (int i = 0; i < 2; ++i)
#pragma unroll
    for (int j = 0; j < 4; ++j) {
      int mBase = bm * BM + wm * 32 + i * 16 + lh * 8;
      int nCol  = bn * BN + wn * 64 + j * 16 + l16;
#pragma unroll
      for (int v = 0; v < 8; ++v) {
        size_t o = (size_t)(mBase + v) * N + nCol;
        if constexpr (OUT_BF16) ((u16*)Cout)[o] = f2bf(acc[i][j][v]);
        else                    ((float*)Cout)[o] = acc[i][j][v];
      }
    }
}

// ---------- flash attention (WMMA, shared-LDS staging) ----------
// grid: B*H*(T/64) blocks, 128 threads (4 waves, 16 t-rows each).
// LDS pool (32KB) reused per phase:
//   phase S1: c chunk   [64][256]            (offset 0, 16384 u16)
//   phase S2: Kp block  [64][128]            (offset 0,  8192 u16)
//   phase PV: P tiles 4x[16][64] (offset 0, 4096) + Vt [128][64] (offset 8192, 8192)

__global__ __launch_bounds__(128) void mla_attn(
    const u16* __restrict__ Qabs,  // [B,T,H,KVL]
    const u16* __restrict__ Qr,    // [B,T,H,HD]
    const u16* __restrict__ Cb,    // [B,T,KVL]
    const u16* __restrict__ Kp,    // [B,KVH,T,HD]
    const u16* __restrict__ Vt,    // [B,KVH,HD,T]
    u16* __restrict__ Ctx)         // [B,T,H,HD]
{
  __shared__ u16 smem[16384];
  int tid = threadIdx.x;
  int lane = tid & 31;
  int wid  = tid >> 5;
  int l16 = lane & 15, lh = lane >> 4;

  int blk   = blockIdx.x;
  int tTile = blk & 31;
  int h     = (blk >> 5) & 15;
  int b     = blk >> 9;
  int g     = h >> 2;                 // rep = H/KVH = 4
  int tW    = tTile * 64 + wid * 16;  // wave's first t row
  const float scale = 1.0f / 16.0f;   // 1/sqrt(2*HD)

  v8f ctxa[8];
#pragma unroll
  for (int nb = 0; nb < 8; ++nb)
#pragma unroll
    for (int v = 0; v < 8; ++v) ctxa[nb][v] = 0.f;
  float rM[8], rS[8];
#pragma unroll
  for (int v = 0; v < 8; ++v) { rM[v] = -3.0e38f; rS[v] = 0.f; }

  size_t qaBase = ((size_t)(b * TT + tW) * HH + h) * KVLAT;
  size_t qrBase = ((size_t)(b * TT + tW) * HH + h) * HDIM;
  const int qaStride = HH * KVLAT;
  const int qrStride = HH * HDIM;
  size_t cRow  = (size_t)b * TT;                       // row base into Cb
  size_t kpRow = (size_t)(b * KVHH + g) * TT;          // row base into Kp
  size_t vtRow = (size_t)(b * KVHH + g) * HDIM;        // row base into Vt

  int nSB = tTile + 1;               // causal: s-blocks 0..tTile
  for (int sb = 0; sb < nSB; ++sb) {
    int sBase = sb * 64;
    v8f sAcc[4];
#pragma unroll
    for (int j = 0; j < 4; ++j)
#pragma unroll
      for (int v = 0; v < 8; ++v) sAcc[j][v] = 0.f;

    // ---- S1 = Qabs * c^T  (K = 512, staged in 2 LDS chunks of 256) ----
#pragma unroll
    for (int half = 0; half < 2; ++half) {
      int kc0 = half * 256;
      __syncthreads();                       // prior phase readers done
      for (int i = tid; i < 2048; i += 128) {        // 64 rows x 32 b128
        int row = i >> 5, kq = (i & 31) * 8;
        async_copy_b128(Cb + (cRow + sBase + row) * KVLAT + kc0 + kq,
                        smem + row * 256 + kq);
      }
      async_wait_all();
      __syncthreads();
      for (int k0 = 0; k0 < 256; k0 += 32) {
        v16bf a = load_a_frag(Qabs + qaBase + kc0 + k0, qaStride, lane);
#pragma unroll
        for (int j = 0; j < 4; ++j) {
          v16bf bfr = load_b_frag(smem + (j * 16) * 256 + k0, 256, lane);
          sAcc[j] = wmma_bf16(a, bfr, sAcc[j]);
        }
      }
    }

    // ---- S2 = Qr * Kp^T  (K = 128, staged in LDS) ----
    __syncthreads();
    for (int i = tid; i < 1024; i += 128) {          // 64 rows x 16 b128
      int row = i >> 4, kq = (i & 15) * 8;
      async_copy_b128(Kp + (kpRow + sBase + row) * HDIM + kq,
                      smem + row * 128 + kq);
    }
    async_wait_all();
    __syncthreads();
    for (int k0 = 0; k0 < HDIM; k0 += 32) {
      v16bf a = load_a_frag(Qr + qrBase + k0, qrStride, lane);
#pragma unroll
      for (int j = 0; j < 4; ++j) {
        v16bf bfr = load_b_frag(smem + (j * 16) * 128 + k0, 128, lane);
        sAcc[j] = wmma_bf16(a, bfr, sAcc[j]);
      }
    }

    // ---- online softmax (row stats per vgpr-slot; rows m = 8*lh + v) ----
#pragma unroll
    for (int v = 0; v < 8; ++v) {
      int t = tW + lh * 8 + v;
      float mx = -3.0e38f;
#pragma unroll
      for (int j = 0; j < 4; ++j) {
        float val = sAcc[j][v] * scale;
        int s = sBase + j * 16 + l16;
        if (s > t) val = -3.0e38f;
        sAcc[j][v] = val;
        mx = fmaxf(mx, val);
      }
#pragma unroll
      for (int off = 8; off >= 1; off >>= 1) mx = fmaxf(mx, __shfl_xor(mx, off, 32));
      float nM = fmaxf(rM[v], mx);
      float cf = __expf(rM[v] - nM);
      rM[v] = nM;
      float ps = 0.f;
#pragma unroll
      for (int j = 0; j < 4; ++j) {
        float p = __expf(sAcc[j][v] - nM);
        sAcc[j][v] = p;
        ps += p;
      }
#pragma unroll
      for (int off = 8; off >= 1; off >>= 1) ps += __shfl_xor(ps, off, 32);
      rS[v] = rS[v] * cf + ps;
#pragma unroll
      for (int nb = 0; nb < 8; ++nb) ctxa[nb][v] *= cf;
    }

    // ---- stage P (C-layout -> row-major 16x64 per wave) and Vt, then P@V ----
    __syncthreads();                   // Kp region readers done
#pragma unroll
    for (int v = 0; v < 8; ++v)
#pragma unroll
      for (int j = 0; j < 4; ++j)
        smem[wid * 1024 + (lh * 8 + v) * 64 + j * 16 + l16] = f2bf(sAcc[j][v]);
    for (int i = tid; i < 1024; i += 128) {          // 128 rows x 8 b128
      int row = i >> 3, kq = (i & 7) * 8;
      async_copy_b128(Vt + (vtRow + row) * TT + sBase + kq,
                      smem + 8192 + row * 64 + kq);
    }
    async_wait_all();
    __syncthreads();
#pragma unroll
    for (int kc = 0; kc < 64; kc += 32) {
      v16bf a = load_a_frag(smem + wid * 1024 + kc, 64, lane);
#pragma unroll
      for (int nb = 0; nb < 8; ++nb) {
        v16bf bfr = load_b_frag(smem + 8192 + (nb * 16) * 64 + kc, 64, lane);
        ctxa[nb] = wmma_bf16(a, bfr, ctxa[nb]);
      }
    }
    __syncthreads();                   // P/Vt readers done before next s-block
  }

  // epilogue: ctx[b,t,h,d] bf16
#pragma unroll
  for (int nb = 0; nb < 8; ++nb)
#pragma unroll
    for (int v = 0; v < 8; ++v) {
      int t = tW + lh * 8 + v;
      int d = nb * 16 + l16;
      float val = ctxa[nb][v] / rS[v];
      Ctx[((size_t)(b * TT + t) * HH + h) * HDIM + d] = f2bf(val);
    }
}

// ---------- host ----------

extern "C" void kernel_launch(void* const* d_in, const int* in_sizes, int n_in,
                              void* d_out, int out_size, void* d_ws, size_t ws_size,
                              hipStream_t stream) {
  (void)in_sizes; (void)n_in; (void)out_size; (void)ws_size;
  const float* x    = (const float*)d_in[0];
  const float* Wdq  = (const float*)d_in[1];
  const float* qnw  = (const float*)d_in[2];
  const float* Wuq  = (const float*)d_in[3];
  const float* Wqr  = (const float*)d_in[4];
  const float* Wdkv = (const float*)d_in[5];
  const float* kvnw = (const float*)d_in[6];
  const float* Wuk  = (const float*)d_in[7];
  const float* Wuv  = (const float*)d_in[8];
  const float* Wkr  = (const float*)d_in[9];
  const float* Wout = (const float*)d_in[10];
  float* out = (float*)d_out;

  char* ws = (char*)d_ws;
  size_t off = 0;
  auto alloc = [&](size_t bytes) -> void* {
    void* p = ws + off;
    off = (off + bytes + 255) & ~(size_t)255;
    return p;
  };
  const size_t M = (size_t)BB * TT;  // 4096

  u16* xb    = (u16*)alloc(M * DDIM * 2);
  u16* wdqb  = (u16*)alloc((size_t)QLAT * DDIM * 2);
  u16* wuqb  = (u16*)alloc((size_t)HH * HDIM * QLAT * 2);
  u16* wqrb  = (u16*)alloc((size_t)HH * HDIM * QLAT * 2);
  u16* wdkvb = (u16*)alloc((size_t)KVLAT * DDIM * 2);
  u16* wukT  = (u16*)alloc((size_t)KVLAT * HDIM * 2);
  u16* wuvb  = (u16*)alloc((size_t)KVHH * HDIM * KVLAT * 2);
  u16* wkrb  = (u16*)alloc((size_t)KVHH * HDIM * DDIM * 2);
  u16* woutb = (u16*)alloc((size_t)DDIM * HH * HDIM * 2);
  float* scr = (float*)alloc(M * 2048 * 4);          // reused fp32 scratch
  u16* qlb   = (u16*)alloc(M * QLAT * 2);
  u16* Qb    = (u16*)alloc(M * HH * HDIM * 2);
  u16* Qrb   = (u16*)alloc(M * HH * HDIM * 2);
  u16* cbuf  = (u16*)alloc(M * KVLAT * 2);
  u16* Qabs  = (u16*)alloc(M * HH * KVLAT * 2);
  u16* Vt    = (u16*)alloc((size_t)BB * KVHH * HDIM * TT * 2);
  u16* Kp    = (u16*)alloc((size_t)BB * KVHH * TT * HDIM * 2);
  u16* ctxb  = (u16*)alloc(M * HH * HDIM * 2);

  auto g1 = [](size_t n) { return (unsigned)((n + 255) / 256); };

  // stage bf16 copies of activations & weights
  cvt_f32_bf16<<<g1(M * DDIM), 256, 0, stream>>>(x, xb, M * DDIM);
  cvt_f32_bf16<<<g1((size_t)QLAT * DDIM), 256, 0, stream>>>(Wdq, wdqb, (size_t)QLAT * DDIM);
  cvt_f32_bf16<<<g1((size_t)HH * HDIM * QLAT), 256, 0, stream>>>(Wuq, wuqb, (size_t)HH * HDIM * QLAT);
  cvt_f32_bf16<<<g1((size_t)HH * HDIM * QLAT), 256, 0, stream>>>(Wqr, wqrb, (size_t)HH * HDIM * QLAT);
  cvt_f32_bf16<<<g1((size_t)KVLAT * DDIM), 256, 0, stream>>>(Wdkv, wdkvb, (size_t)KVLAT * DDIM);
  cvt_T_bf16<<<g1((size_t)HDIM * KVLAT), 256, 0, stream>>>(Wuk, wukT, HDIM, KVLAT);
  cvt_f32_bf16<<<g1((size_t)KVHH * HDIM * KVLAT), 256, 0, stream>>>(Wuv, wuvb, (size_t)KVHH * HDIM * KVLAT);
  cvt_f32_bf16<<<g1((size_t)KVHH * HDIM * DDIM), 256, 0, stream>>>(Wkr, wkrb, (size_t)KVHH * HDIM * DDIM);
  cvt_f32_bf16<<<g1((size_t)DDIM * HH * HDIM), 256, 0, stream>>>(Wout, woutb, (size_t)DDIM * HH * HDIM);

  // ql = rms(x @ Wdq^T)
  gemm_bf16_nt<false><<<dim3(32, QLAT / BN), 256, 0, stream>>>(xb, wdqb, scr, (int)M, QLAT, DDIM);
  rms_norm_bf16<<<(unsigned)M, 256, 0, stream>>>(scr, qnw, qlb, QLAT);
  // Q = ql @ Wuq^T  (bf16 direct)
  gemm_bf16_nt<true><<<dim3(32, 16), 256, 0, stream>>>(qlb, wuqb, Qb, (int)M, HH * HDIM, QLAT);
  // Qr = rope_t(ql @ Wqr^T)
  gemm_bf16_nt<false><<<dim3(32, 16), 256, 0, stream>>>(qlb, wqrb, scr, (int)M, HH * HDIM, QLAT);
  rope_t_bf16<<<g1((size_t)BB * TT * HH * (HDIM / 2)), 256, 0, stream>>>(scr, Qrb);
  // c = rms(x @ Wdkv^T)
  gemm_bf16_nt<false><<<dim3(32, 4), 256, 0, stream>>>(xb, wdkvb, scr, (int)M, KVLAT, DDIM);
  rms_norm_bf16<<<(unsigned)M, 256, 0, stream>>>(scr, kvnw, cbuf, KVLAT);
  // Qabs = Q @ Wuk  (M = B*T*H, K = HD, via Wuk^T)
  gemm_bf16_nt<true><<<dim3(512, 4), 256, 0, stream>>>(Qb, wukT, Qabs, (int)(M * HH), KVLAT, HDIM);
  // V = c @ Wuv^T -> transpose to [B,KVH,HD,T]
  gemm_bf16_nt<false><<<dim3(32, 4), 256, 0, stream>>>(cbuf, wuvb, scr, (int)M, KVHH * HDIM, KVLAT);
  v_trans_bf16<<<g1((size_t)BB * TT * KVLAT), 256, 0, stream>>>(scr, Vt);
  // Kp = rope_h(x @ Wkr^T) -> [B,KVH,T,HD]
  gemm_bf16_nt<false><<<dim3(32, 4), 256, 0, stream>>>(xb, wkrb, scr, (int)M, KVHH * HDIM, DDIM);
  rope_h_bf16<<<g1((size_t)BB * TT * KVHH * (HDIM / 2)), 256, 0, stream>>>(scr, Kp);
  // flash attention -> ctx bf16 [B,T,H,HD]
  mla_attn<<<BB * HH * (TT / 64), 128, 0, stream>>>(Qabs, Qrb, cbuf, Kp, Vt, ctxb);
  // out = ctx @ Wout^T  (fp32)
  gemm_bf16_nt<false><<<dim3(32, 16), 256, 0, stream>>>(ctxb, woutb, out, (int)M, DDIM, HH * HDIM);
}
